// GlobalAttention_17145509446429
// MI455X (gfx1250) — compile-verified, matrix operand order checked by reading, not run on hardware
//
#include <hip/hip_runtime.h>
#include <hip/hip_bf16.h>

// ---------------------------------------------------------------------------
// GlobalAttention (Luong "general") for MI455X / gfx1250, wave32 + WMMA bf16.
//   B=16, T=1024, S=1536, Dq=Dk=1024
// Pipeline: (0) convert fp32 operands -> bf16 in ws (once)
//           (1) q_proj = q16 @ Win16^T              -> bf16 ws
//           (2) align  = qproj16 @ mem16^T          -> fp32 d_out (raw)
//           (3) masked softmax: align(masked) fp32, probs fp32 (d_out) + bf16 (ws)
//           (4) c      = probs16 @ mem16            -> bf16 ws
//           (5) attn_h = tanh(concat(c16,q16) @ Wout16^T) -> fp32 d_out
// GEMMs: 128x256 block tile, 8 waves, 64x64 wave tile (16 WMMA / K-step),
// double-buffered LDS, async global->LDS copies (ASYNCcnt) for streaming.
// ws usage ~198 MB.
// ---------------------------------------------------------------------------

typedef __attribute__((ext_vector_type(16))) __bf16 v16bf;
typedef __attribute__((ext_vector_type(8)))  __bf16 v8bf;
typedef __attribute__((ext_vector_type(4)))  __bf16 v4bf;
typedef __attribute__((ext_vector_type(8)))  float  v8f;
typedef int v4i __attribute__((vector_size(16)));

#define BM 128        // block tile M
#define BNC 256       // block tile N
#define BK 32         // K step (one WMMA K)
#define LDT 40        // LDS row stride in bf16 (BK + 8 pad -> 80 B rows)

#if defined(__gfx1250__) && __has_builtin(__builtin_amdgcn_global_load_async_to_lds_b128)
#define HAS_ASYNC 1
#else
#define HAS_ASYNC 0
#endif

typedef __attribute__((address_space(1))) v4i* gv4i_p;   // global int4*
typedef __attribute__((address_space(3))) v4i* lv4i_p;   // LDS int4*

__device__ __forceinline__ void cp_b128(const __bf16* g, __bf16* l) {
#if HAS_ASYNC
    __builtin_amdgcn_global_load_async_to_lds_b128(
        (gv4i_p)const_cast<__bf16*>(g), (lv4i_p)l, 0, 0);
#else
    *reinterpret_cast<uint4*>(l) = *reinterpret_cast<const uint4*>(g);
#endif
}

__device__ __forceinline__ void async_wait0() {
#if HAS_ASYNC
#if __has_builtin(__builtin_amdgcn_s_wait_asynccnt)
    __builtin_amdgcn_s_wait_asynccnt(0);
#else
    asm volatile("s_wait_asynccnt 0" ::: "memory");
#endif
#endif
}

// C = op(A_concat) * op(B); A/B bf16, accumulate f32, output TC (float or bf16).
//  B_NK   : B stored row-major [N,K] (compute A*B_storage^T); else [K,N]
//  DO_TANH: tanh epilogue
//  CONCAT : A columns [0,ksplit) from A, [ksplit,K) from A2
template<bool B_NK, bool DO_TANH, bool CONCAT, typename TC>
__global__ __launch_bounds__(256)
void gemm_bf16_wmma(const __bf16* __restrict__ A,
                    const __bf16* __restrict__ A2, int ksplit,
                    const __bf16* __restrict__ B,
                    TC* __restrict__ C,
                    int M, int N, int K,
                    long long strideA, long long strideB, long long strideC)
{
    __shared__ __align__(16) __bf16 sA[2][BM * LDT];
    __shared__ __align__(16) __bf16 sB[2][BNC * LDT];

    const int z = blockIdx.z;
    A += (long long)z * strideA;
    B += (long long)z * strideB;
    C += (long long)z * strideC;

    const int tid  = threadIdx.x;
    const int wave = tid >> 5;
    const int lane = tid & 31;
    const int l    = lane & 15;          // row/col within 16
    const int hi   = lane >> 4;          // K-half select
    const int wm   = (wave >> 2) * 64;   // wave M offset (2 waves in M)
    const int wn   = (wave & 3)  * 64;   // wave N offset (4 waves in N)
    const int m0   = blockIdx.x * BM;
    const int n0   = blockIdx.y * BNC;

    v8f acc[4][4] = {};                  // 64x64 wave tile

    auto stage = [&](int buf, int k0) {
        // ---- A tile: BM x BK bf16, contiguous 64 B rows -> async-capable ----
        const __bf16* Asrc = A;
        int kc  = k0;
        int lda = CONCAT ? ksplit : K;
        if (CONCAT && k0 >= ksplit) { Asrc = A2; kc = k0 - ksplit; lda = K - ksplit; }
#pragma unroll
        for (int it = 0; it < 2; ++it) {
            int idx = tid + it * 256;            // 512 x 16B chunks
            int r = idx >> 2, c = idx & 3;
            cp_b128(Asrc + (long long)(m0 + r) * lda + kc + c * 8,
                    &sA[buf][r * LDT + c * 8]);
        }
        if (B_NK) {
            // ---- B tile: BNC x BK bf16, contiguous rows -> async-capable ----
#pragma unroll
            for (int it = 0; it < 4; ++it) {
                int idx = tid + it * 256;        // 1024 x 16B chunks
                int r = idx >> 2, c = idx & 3;
                cp_b128(B + (long long)(n0 + r) * K + k0 + c * 8,
                        &sB[buf][r * LDT + c * 8]);
            }
        } else {
            // ---- B tile: BK x BNC, transpose into [n][k] (manual path) ----
#pragma unroll
            for (int it = 0; it < 4; ++it) {
                int idx = tid + it * 256;        // 1024 x 8-elem vectors
                int r = idx >> 5, c8 = idx & 31;
                const __bf16* g = B + (long long)(k0 + r) * N + n0 + c8 * 8;
#pragma unroll
                for (int j = 0; j < 8; ++j)
                    sB[buf][(c8 * 8 + j) * LDT + r] = g[j];
            }
        }
    };

    auto compute = [&](int buf) {
        v16bf aF[4], bF[4];
#pragma unroll
        for (int i = 0; i < 4; ++i) {
            const v8bf* p = reinterpret_cast<const v8bf*>(
                &sA[buf][(wm + i * 16 + l) * LDT + hi * 8]);
            v8bf lo = p[0], hv = p[2];           // K={hi*8..}+{hi*8+16..}
#pragma unroll
            for (int e = 0; e < 8; ++e) { aF[i][e] = lo[e]; aF[i][8 + e] = hv[e]; }
        }
#pragma unroll
        for (int j = 0; j < 4; ++j) {
            const v8bf* p = reinterpret_cast<const v8bf*>(
                &sB[buf][(wn + j * 16 + l) * LDT + hi * 8]);
            v8bf lo = p[0], hv = p[2];
#pragma unroll
            for (int e = 0; e < 8; ++e) { bF[j][e] = lo[e]; bF[j][8 + e] = hv[e]; }
        }
#pragma unroll
        for (int i = 0; i < 4; ++i)
#pragma unroll
            for (int j = 0; j < 4; ++j)
                acc[i][j] = __builtin_amdgcn_wmma_f32_16x16x32_bf16(
                    false, aF[i], false, bF[j], (short)0, acc[i][j], false, false);
    };

    // ---- double-buffered main loop: one barrier per K-step ----
    int buf = 0;
    stage(0, 0);
    async_wait0();
    __syncthreads();
    for (int k0 = 0; k0 < K; k0 += BK) {
        int nk = k0 + BK;
        if (nk < K) stage(buf ^ 1, nk);          // stream next tile (async)
        compute(buf);                             // 16 WMMA from current tile
        if (nk < K) { async_wait0(); __syncthreads(); }
        buf ^= 1;
    }

    // ---- store: element e of v8f is row (hi*8+e), col l ----
#pragma unroll
    for (int i = 0; i < 4; ++i)
#pragma unroll
        for (int j = 0; j < 4; ++j)
#pragma unroll
            for (int e = 0; e < 8; ++e) {
                int m = m0 + wm + i * 16 + hi * 8 + e;
                int n = n0 + wn + j * 16 + l;
                float v = acc[i][j][e];
                if (DO_TANH) v = tanhf(v);
                C[(long long)m * N + n] = (TC)v;
            }
}

// ---------------------------------------------------------------------------
// fp32 -> bf16 bulk conversion (vec4), n divisible by 4.
// ---------------------------------------------------------------------------
__global__ __launch_bounds__(256)
void cvt_f32_bf16(const float* __restrict__ s, __bf16* __restrict__ d, int n4)
{
    int i = blockIdx.x * 256 + threadIdx.x;
    if (i < n4) {
        float4 v = reinterpret_cast<const float4*>(s)[i];
        v4bf o = { (__bf16)v.x, (__bf16)v.y, (__bf16)v.z, (__bf16)v.w };
        reinterpret_cast<v4bf*>(d)[i] = o;
    }
}

// ---------------------------------------------------------------------------
// Masked softmax over S=1536. One block per (b,t) row. Writes masked scores
// back to `align`, probs to fp32 d_out region and bf16 ws copy.
// ---------------------------------------------------------------------------
__global__ __launch_bounds__(256)
void softmax_mask_kernel(float* __restrict__ align, float* __restrict__ probs,
                         __bf16* __restrict__ probs16,
                         const int* __restrict__ lens, int T, int S)
{
    constexpr int PER = 6;                 // 1536 / 256
    const int row = blockIdx.x;            // 0 .. B*T-1
    const int b   = row / T;
    const int len = lens[b];
    float*  arow = align   + (long long)row * S;
    float*  prow = probs   + (long long)row * S;
    __bf16* hrow = probs16 + (long long)row * S;
    const int tid = threadIdx.x;

    float vals[PER];
    float mx = -1e30f;
#pragma unroll
    for (int i = 0; i < PER; ++i) {
        int j = tid + i * 256;
        float x = arow[j];
        x = (j < len) ? x : -1e30f;
        vals[i] = x;
        mx = fmaxf(mx, x);
    }
    for (int off = 16; off > 0; off >>= 1) mx = fmaxf(mx, __shfl_xor(mx, off, 32));
    __shared__ float red[8];
    if ((tid & 31) == 0) red[tid >> 5] = mx;
    __syncthreads();
    mx = red[0];
#pragma unroll
    for (int i = 1; i < 8; ++i) mx = fmaxf(mx, red[i]);
    __syncthreads();

    float s = 0.f;
#pragma unroll
    for (int i = 0; i < PER; ++i) {
        float e = expf(vals[i] - mx);      // masked -> exp(-huge) = 0
        arow[tid + i * 256] = vals[i];     // masked scores (output 3)
        vals[i] = e;
        s += e;
    }
    for (int off = 16; off > 0; off >>= 1) s += __shfl_xor(s, off, 32);
    if ((tid & 31) == 0) red[tid >> 5] = s;
    __syncthreads();
    s = red[0];
#pragma unroll
    for (int i = 1; i < 8; ++i) s += red[i];
    float inv = 1.0f / s;
#pragma unroll
    for (int i = 0; i < PER; ++i) {
        float p = vals[i] * inv;
        prow[tid + i * 256] = p;
        hrow[tid + i * 256] = (__bf16)p;
    }
}

// ---------------------------------------------------------------------------
extern "C" void kernel_launch(void* const* d_in, const int* in_sizes, int n_in,
                              void* d_out, int out_size, void* d_ws, size_t ws_size,
                              hipStream_t stream)
{
    (void)in_sizes; (void)n_in; (void)out_size; (void)ws_size;

    constexpr int B = 16, T = 1024, S = 1536, D = 1024;   // Dq = Dk = D
    constexpr long long nQ  = (long long)B * T * D;       // query / qproj / c
    constexpr long long nM  = (long long)B * S * D;       // memory bank / probs
    constexpr long long nWi = (long long)D * D;
    constexpr long long nWo = (long long)D * 2 * D;

    const float* query = (const float*)d_in[0];
    const float* mem   = (const float*)d_in[1];
    const int*   lens  = (const int*)  d_in[2];
    const float* Win   = (const float*)d_in[3];
    const float* Wout  = (const float*)d_in[4];

    float* attn_h = (float*)d_out;                 // [B,T,D]
    float* probs  = attn_h + nQ;                   // [B,T,S]
    float* align  = probs  + (long long)B * T * S; // [B,T,S]

    // bf16 workspace layout
    __bf16* q16     = (__bf16*)d_ws;
    __bf16* m16     = q16     + nQ;
    __bf16* Wi16    = m16     + nM;
    __bf16* Wo16    = Wi16    + nWi;
    __bf16* qproj16 = Wo16    + nWo;
    __bf16* probs16 = qproj16 + nQ;
    __bf16* c16     = probs16 + nM;

    // 0) one-shot operand conversion to bf16
    cvt_f32_bf16<<<(int)(nQ  / 4 + 255) / 256, 256, 0, stream>>>(query, q16,  (int)(nQ  / 4));
    cvt_f32_bf16<<<(int)(nM  / 4 + 255) / 256, 256, 0, stream>>>(mem,   m16,  (int)(nM  / 4));
    cvt_f32_bf16<<<(int)(nWi / 4 + 255) / 256, 256, 0, stream>>>(Win,   Wi16, (int)(nWi / 4));
    cvt_f32_bf16<<<(int)(nWo / 4 + 255) / 256, 256, 0, stream>>>(Wout,  Wo16, (int)(nWo / 4));

    // 1) q_proj = query @ W_in^T        [16384 x 1024] x [1024 x 1024] -> bf16
    {
        dim3 g((B * T) / BM, D / BNC, 1);
        gemm_bf16_wmma<true, false, false, __bf16><<<g, 256, 0, stream>>>(
            q16, nullptr, 0, Wi16, qproj16, B * T, D, D, 0, 0, 0);
    }
    // 2) align_raw = q_proj @ mem^T     batched [1024x1024] x [1024x1536] -> fp32
    {
        dim3 g(T / BM, S / BNC, B);
        gemm_bf16_wmma<true, false, false, float><<<g, 256, 0, stream>>>(
            qproj16, nullptr, 0, m16, align, T, S, D,
            (long long)T * D, (long long)S * D, (long long)T * S);
    }
    // 3) mask + softmax
    softmax_mask_kernel<<<B * T, 256, 0, stream>>>(align, probs, probs16, lens, T, S);

    // 4) c = probs @ mem                batched [1024x1536] x [1536x1024] -> bf16
    {
        dim3 g(T / BM, D / BNC, B);
        gemm_bf16_wmma<false, false, false, __bf16><<<g, 256, 0, stream>>>(
            probs16, nullptr, 0, m16, c16, T, D, S,
            (long long)T * S, (long long)S * D, (long long)T * D);
    }
    // 5) attn_h = tanh(concat(c, query) @ W_out^T)  [16384x2048] x [2048x1024]
    {
        dim3 g((B * T) / BM, D / BNC, 1);
        gemm_bf16_wmma<true, true, true, float><<<g, 256, 0, stream>>>(
            c16, q16, D, Wo16, attn_h, B * T, D, 2 * D, 0, 0, 0);
    }
}
